// Trainer_14465449853585
// MI455X (gfx1250) — compile-verified, matrix operand order checked by reading, not run on hardware
//
#include <hip/hip_runtime.h>

// ---------------------------------------------------------------------------
// MI455X (gfx1250) fused  normalize -> GEMM(bf16-split WMMA) -> softmax@label
//
//   f      = x / ||x||                      [B, D]   B=4096, D=256
//   logits = (f @ centrals^T) / TEMP        [B, C]   C=32768, TEMP=0.05
//   out[b] = softmax(logits[b])[labels[b]]  [B]
//
// Design:
//  * fp32 -> bf16 hi/lo split; 3 bf16 WMMAs (hi*hi + hi*lo + lo*hi) per K-chunk
//    gives near-fp32 accuracy at bf16 matrix-engine rate (~3x faster than the
//    f32 16x16x4 WMMA path for the 68.7 GFLOP GEMM).
//  * |logit| <= 1/TEMP = 20, so sum(exp) is fp32-safe without max-subtraction:
//    streaming softmax fuses into the GEMM epilogue; the 512MB logits matrix
//    never exists (saves ~1GB of HBM traffic ~= 44us at 23.3 TB/s).
//  * Row-norm and 1/TEMP folded into the A-side bf16 conversion.
//  * Per block: 32 rows of A staged in LDS via per-lane async global->LDS
//    copies (ASYNCcnt path); all 32768 columns streamed; bf16 centrals
//    (32MB hi+lo) stay L2-resident (192MB L2) across the 128 blocks.
// ---------------------------------------------------------------------------

typedef __attribute__((ext_vector_type(16))) __bf16 v16bf;
typedef __attribute__((ext_vector_type(8)))  __bf16 v8bf;
typedef __attribute__((ext_vector_type(4)))  __bf16 v4bf;
typedef __attribute__((ext_vector_type(8)))  float  v8f;
typedef __attribute__((ext_vector_type(4)))  float  v4f;

// GCC-style vector type matching the async-to-LDS builtin's pointee type
// ("__vector_size__(4*sizeof(int)) int" per the clang diagnostic).
typedef int v4i_gcc __attribute__((vector_size(16)));
typedef __attribute__((address_space(1))) v4i_gcc* async_gptr;
typedef __attribute__((address_space(3))) v4i_gcc* async_lptr;

#define B_ROWS 4096
#define C_COLS 32768
#define D_K    256
#define INV_TEMP 20.0f

// round-to-nearest-even fp32 -> bf16 (bit manipulation: no dependence on
// native cvt availability)
__device__ __forceinline__ __bf16 f2bf(float f) {
    unsigned u = __builtin_bit_cast(unsigned, f);
    unsigned r = (u + 0x7FFFu + ((u >> 16) & 1u)) >> 16;
    unsigned short s = (unsigned short)r;
    return __builtin_bit_cast(__bf16, s);
}
__device__ __forceinline__ float bf2f(__bf16 b) {
    unsigned short s = __builtin_bit_cast(unsigned short, b);
    unsigned u = ((unsigned)s) << 16;
    return __builtin_bit_cast(float, u);
}

// Load one 16-bit WMMA fragment for this lane from a row-major bf16 row
// pointer already offset to (row, k0) where k0 = (lane>=16 ? 8 : 0) + kchunk*32.
// Per ISA layout: halves 0-7 = K[k0..k0+7], halves 8-15 = K[k0+16..k0+23].
__device__ __forceinline__ v16bf load_frag(const __bf16* p) {
    v8bf a = *(const v8bf*)(p);
    v8bf b = *(const v8bf*)(p + 16);
    return __builtin_shufflevector(a, b, 0,1,2,3,4,5,6,7,8,9,10,11,12,13,14,15);
}

// ---------------------------------------------------------------------------
// Kernel 0: per-row L2 norm of x, fold rsqrt * (1/TEMP) into the value, and
// emit bf16 hi/lo split.  One wave per row (wave32): lane owns 8 elements.
// ---------------------------------------------------------------------------
__global__ __launch_bounds__(256) void prep_x_kernel(
    const float* __restrict__ x,
    __bf16* __restrict__ Ahi, __bf16* __restrict__ Alo)
{
    const int wave = threadIdx.x >> 5;
    const int lane = threadIdx.x & 31;
    const int row  = blockIdx.x * 8 + wave;

    const float* xr = x + (size_t)row * D_K + lane * 8;
    v4f a = *(const v4f*)(xr);
    v4f b = *(const v4f*)(xr + 4);

    float ss = a[0]*a[0] + a[1]*a[1] + a[2]*a[2] + a[3]*a[3]
             + b[0]*b[0] + b[1]*b[1] + b[2]*b[2] + b[3]*b[3];
    ss += __shfl_xor(ss, 1,  32);
    ss += __shfl_xor(ss, 2,  32);
    ss += __shfl_xor(ss, 4,  32);
    ss += __shfl_xor(ss, 8,  32);
    ss += __shfl_xor(ss, 16, 32);
    const float s = rsqrtf(ss) * INV_TEMP;   // fold normalize + 1/TEMP

    v8bf vh, vl;
    #pragma unroll
    for (int i = 0; i < 8; ++i) {
        float v = (i < 4 ? a[i] : b[i - 4]) * s;
        __bf16 h = f2bf(v);
        vh[i] = h;
        vl[i] = f2bf(v - bf2f(h));
    }
    *(v8bf*)(Ahi + (size_t)row * D_K + lane * 8) = vh;
    *(v8bf*)(Alo + (size_t)row * D_K + lane * 8) = vl;
}

// ---------------------------------------------------------------------------
// Kernel 1: centrals fp32 -> bf16 hi/lo split (elementwise, float4 per thread)
// ---------------------------------------------------------------------------
__global__ __launch_bounds__(256) void prep_c_kernel(
    const float* __restrict__ c,
    __bf16* __restrict__ Bhi, __bf16* __restrict__ Blo)
{
    const int i = blockIdx.x * blockDim.x + threadIdx.x;   // float4 index
    v4f v = *(const v4f*)(c + (size_t)i * 4);
    v4bf vh, vl;
    #pragma unroll
    for (int k = 0; k < 4; ++k) {
        __bf16 h = f2bf(v[k]);
        vh[k] = h;
        vl[k] = f2bf(v[k] - bf2f(h));
    }
    *(v4bf*)(Bhi + (size_t)i * 4) = vh;
    *(v4bf*)(Blo + (size_t)i * 4) = vl;
}

// ---------------------------------------------------------------------------
// Kernel 2: fused GEMM + streaming sum(exp) + label-logit capture.
// Block = 256 threads = 8 waves arranged 2(M) x 4(N); block owns MT=32 rows
// for ALL of C.  Each wave computes 16x16 logit tiles: K=256 in 8 chunks of
// 32, 3 bf16 WMMAs per chunk (hi*hi, hi*lo, lo*hi).
// ---------------------------------------------------------------------------
#define MT   32
#define LDA  272   // padded LDS row stride (elements) to spread banks

__global__ __launch_bounds__(256) void gemm_softmax_kernel(
    const __bf16* __restrict__ Ahi, const __bf16* __restrict__ Alo,
    const __bf16* __restrict__ Bhi, const __bf16* __restrict__ Blo,
    const int* __restrict__ labels, float* __restrict__ out)
{
    __shared__ __align__(16) __bf16 sAhi[MT * LDA];
    __shared__ __align__(16) __bf16 sAlo[MT * LDA];
    __shared__ float sSum[MT];
    __shared__ float sNum[MT];

    const int tid  = threadIdx.x;
    const int wave = tid >> 5;
    const int lane = tid & 31;
    const int wm   = wave >> 2;     // 0..1 : M sub-block (16 rows)
    const int wn   = wave & 3;      // 0..3 : N sub-block (16 cols)
    const int row_base = blockIdx.x * MT;

    if (tid < MT) { sSum[tid] = 0.0f; sNum[tid] = 0.0f; }

    // Stage the 32-row A tile (hi & lo) into padded LDS, 16B chunks per lane.
#if __has_builtin(__builtin_amdgcn_global_load_async_to_lds_b128)
    // CDNA5 async global->LDS path: data bypasses VGPRs, tracked on ASYNCcnt.
    for (int cidx = tid; cidx < MT * (D_K / 8); cidx += 256) {
        const int r  = cidx >> 5;            // D_K/8 = 32 chunks per row
        const int k8 = (cidx & 31) * 8;
        __builtin_amdgcn_global_load_async_to_lds_b128(
            (async_gptr)&Ahi[(size_t)(row_base + r) * D_K + k8],
            (async_lptr)&sAhi[r * LDA + k8],
            0, 0);
        __builtin_amdgcn_global_load_async_to_lds_b128(
            (async_gptr)&Alo[(size_t)(row_base + r) * D_K + k8],
            (async_lptr)&sAlo[r * LDA + k8],
            0, 0);
    }
#if __has_builtin(__builtin_amdgcn_s_wait_asynccnt)
    __builtin_amdgcn_s_wait_asynccnt(0);
#else
    asm volatile("s_wait_asynccnt 0x0" ::: "memory");
#endif
#else
    for (int cidx = tid; cidx < MT * (D_K / 8); cidx += 256) {
        const int r  = cidx >> 5;
        const int k8 = (cidx & 31) * 8;
        *(v8bf*)(&sAhi[r * LDA + k8]) =
            *(const v8bf*)(&Ahi[(size_t)(row_base + r) * D_K + k8]);
        *(v8bf*)(&sAlo[r * LDA + k8]) =
            *(const v8bf*)(&Alo[(size_t)(row_base + r) * D_K + k8]);
    }
#endif
    __syncthreads();

    const int half = lane >> 4;      // K-phase of this lane in the fragment
    const int l16  = lane & 15;

    // Labels for the 8 rows this lane's acc elements belong to (j = 0..7:
    // C/D layout puts M = j + 8*half in VGPR j).
    int lab[8];
    #pragma unroll
    for (int j = 0; j < 8; ++j)
        lab[j] = labels[row_base + wm * 16 + half * 8 + j];

    float psum[8];
    #pragma unroll
    for (int j = 0; j < 8; ++j) psum[j] = 0.0f;

    // LDS A fragment base for this lane: row = wm*16 + l16, k0 = half*8
    const __bf16* sArowH = &sAhi[(wm * 16 + l16) * LDA + half * 8];
    const __bf16* sArowL = &sAlo[(wm * 16 + l16) * LDA + half * 8];

    for (int col0 = wn * 16; col0 < C_COLS; col0 += 64) {
        const int n = col0 + l16;    // this lane's column = centrals row
        const __bf16* bH = &Bhi[(size_t)n * D_K + half * 8];
        const __bf16* bL = &Blo[(size_t)n * D_K + half * 8];

        v8f acc = {};
        #pragma unroll
        for (int kc = 0; kc < 8; ++kc) {
            v16bf aH = load_frag(sArowH + kc * 32);
            v16bf aL = load_frag(sArowL + kc * 32);
            v16bf fH = load_frag(bH + kc * 32);
            v16bf fL = load_frag(bL + kc * 32);
            acc = __builtin_amdgcn_wmma_f32_16x16x32_bf16(
                      false, aH, false, fH, (short)0, acc, false, false);
            acc = __builtin_amdgcn_wmma_f32_16x16x32_bf16(
                      false, aH, false, fL, (short)0, acc, false, false);
            acc = __builtin_amdgcn_wmma_f32_16x16x32_bf16(
                      false, aL, false, fH, (short)0, acc, false, false);
        }

        // Prefetch next iteration's B rows (L2-resident; warms WGP$/L0).
        if (col0 + 64 < C_COLS) {
            __builtin_prefetch(&Bhi[(size_t)(n + 64) * D_K], 0, 3);
            __builtin_prefetch(&Blo[(size_t)(n + 64) * D_K], 0, 3);
        }

        // Epilogue: |logit| <= 20 so exp is fp32-safe without max-shift.
        #pragma unroll
        for (int j = 0; j < 8; ++j) {
            const float logit = acc[j];
            psum[j] += __expf(logit);
            if (n == lab[j])
                sNum[wm * 16 + half * 8 + j] = logit;   // unique writer
        }
    }

    // Reduce psum across the 16 lanes sharing each row, then combine the
    // 4 N-waves via LDS float atomics.
    #pragma unroll
    for (int j = 0; j < 8; ++j) {
        float v = psum[j];
        v += __shfl_xor(v, 1, 32);
        v += __shfl_xor(v, 2, 32);
        v += __shfl_xor(v, 4, 32);
        v += __shfl_xor(v, 8, 32);
        if (l16 == 0)
            atomicAdd(&sSum[wm * 16 + half * 8 + j], v);   // ds_add_f32
    }
    __syncthreads();

    if (tid < MT)
        out[row_base + tid] = __expf(sNum[tid]) / sSum[tid];
}

// ---------------------------------------------------------------------------
// Launcher
// ---------------------------------------------------------------------------
extern "C" void kernel_launch(void* const* d_in, const int* in_sizes, int n_in,
                              void* d_out, int out_size, void* d_ws, size_t ws_size,
                              hipStream_t stream)
{
    const float* x        = (const float*)d_in[0];   // [4096, 256]
    const float* centrals = (const float*)d_in[1];   // [32768, 256]
    const int*   labels   = (const int*)d_in[2];     // [4096]
    float*       out      = (float*)d_out;           // [4096]

    // Workspace layout (bf16 hi/lo splits): 2MB + 2MB + 16MB + 16MB = 36MB
    __bf16* Ahi = (__bf16*)d_ws;
    __bf16* Alo = Ahi + (size_t)B_ROWS * D_K;
    __bf16* Bhi = Alo + (size_t)B_ROWS * D_K;
    __bf16* Blo = Bhi + (size_t)C_COLS * D_K;

    prep_x_kernel<<<B_ROWS / 8, 256, 0, stream>>>(x, Ahi, Alo);
    prep_c_kernel<<<(C_COLS * D_K / 4) / 256, 256, 0, stream>>>(centrals, Bhi, Blo);
    gemm_softmax_kernel<<<B_ROWS / MT, 256, 0, stream>>>(Ahi, Alo, Bhi, Blo,
                                                         labels, out);
}
